// BLRU_82154134438695
// MI455X (gfx1250) — matrix operand order repaired and checked by reading, not compiled, not run on hardware
//
#include <hip/hip_runtime.h>
#include <hip/hip_bf16.h>
#include <math.h>

// ---------------------------------------------------------------------------
// Bidirectional LRU (diagonal complex linear recurrence) for MI455X / gfx1250.
//   1) cvt u -> bf16;  build W1t = [Bc.re | Bc.im]^T (1024x512 bf16, [n][k]),
//      W2t = [C0 ; -C1]^T (512x1024 bf16, [j][k]), lam = exp(-exp(nu)+i exp(th))
//   2) WMMA bf16 GEMM:  Bu(32768x1024 f32) = uB(32768x512) @ W1
//   3) blocked scan over L (64 segs x 64 steps), fwd n<256 / bwd n>=256,
//      in-place on Bu, lam^64 cross-segment carry fix, masked bf16 X out
//   4) WMMA bf16 GEMM:  y(32768x512 f32) = X @ W2
// GEMM uses double-buffered LDS fed by GLOBAL_LOAD_ASYNC_TO_LDS_B128 when the
// toolchain exposes it (ASYNCcnt pipeline), else a register-staged fallback.
// ---------------------------------------------------------------------------

typedef __attribute__((ext_vector_type(16))) __bf16 v16bf;
typedef __attribute__((ext_vector_type(8)))  float  v8f;
typedef __attribute__((ext_vector_type(4)))  int    v4i;

#define H_DIM  512
#define N_DIM  512
#define BSZ    8
#define LSEQ   4096
#define MROWS  (BSZ * LSEQ)      // 32768
#define SEGS   64
#define SEGLEN (LSEQ / SEGS)     // 64

#if __has_builtin(__builtin_amdgcn_global_load_async_to_lds_b128) && \
    __has_builtin(__builtin_amdgcn_s_wait_asynccnt)
#define USE_ASYNC_LDS 1
#endif

static __device__ __forceinline__ __bf16 f2bf(float f) { return (__bf16)f; }

#ifdef USE_ASYNC_LDS
static __device__ __forceinline__ void blru_async_cp16(void* lds, const void* g) {
    // builtin signature: (v4i addrspace(1)*, v4i addrspace(3)*, imm offset, imm cpol)
    __builtin_amdgcn_global_load_async_to_lds_b128(
        (__attribute__((address_space(1))) v4i*)(g),
        (__attribute__((address_space(3))) v4i*)(lds),
        0, 0);
}
#endif

// ---------------------------------------------------------------- parameters
__global__ void blru_lam_kernel(const float* __restrict__ nu_log,
                                const float* __restrict__ theta_log,
                                float* __restrict__ lam_re,
                                float* __restrict__ lam_im) {
    int n = blockIdx.x * blockDim.x + threadIdx.x;
    if (n < N_DIM) {
        float mag = expf(-expf(nu_log[n]));
        float ph  = expf(theta_log[n]);
        lam_re[n] = mag * cosf(ph);
        lam_im[n] = mag * sinf(ph);
    }
}

// W1t[n][k] = Bc[k][n].re (n<512) | Bc[k][n-512].im  -- pre-transposed [n][k]
__global__ void blru_w1_kernel(const float* __restrict__ Bin, __bf16* __restrict__ W1t) {
    int idx = blockIdx.x * 256 + threadIdx.x;          // < 1024*512
    int n = idx >> 9, k = idx & 511;
    float v = (n < 512) ? Bin[(k * 512 + n) * 2]
                        : Bin[(k * 512 + (n - 512)) * 2 + 1];
    W1t[idx] = f2bf(v);
}

// W2t[j][k] = C0[k][j] (k<512) | -C1[k-512][j]  -- pre-transposed [j][k]
__global__ void blru_w2_kernel(const float* __restrict__ Cin, __bf16* __restrict__ W2t) {
    int idx = blockIdx.x * 256 + threadIdx.x;          // < 512*1024
    int j = idx >> 10, k = idx & 1023;
    float v = (k < 512) ? Cin[k * 512 + j]
                        : -Cin[512 * 512 + (k - 512) * 512 + j];
    W2t[idx] = f2bf(v);
}

__global__ void blru_cvtu_kernel(const float* __restrict__ u, __bf16* __restrict__ uB, int n4) {
    int i = blockIdx.x * blockDim.x + threadIdx.x;
    if (i < n4) {
        float4 f = ((const float4*)u)[i];
        uB[i * 4 + 0] = f2bf(f.x);
        uB[i * 4 + 1] = f2bf(f.y);
        uB[i * 4 + 2] = f2bf(f.z);
        uB[i * 4 + 3] = f2bf(f.w);
    }
}

// -------------------------------------------------------------- WMMA GEMM
// Row-major A (MxK) bf16  x  row-major-transposed Bt (NxK) bf16 -> Out (MxN) f32.
// Block: 256 threads (8 waves), tile 128(M) x 128(N), K-step 32, double-buffered
// LDS staged via async global->LDS copies. Wave w: rows (w&3)*32, cols (w>>2)*64.
#define BM  128
#define BN  128
#define BK  32
#define LDT 48   // padded LDS stride (keeps 16B alignment, spreads banks)

__global__ __launch_bounds__(256) void blru_gemm_kernel(
        const __bf16* __restrict__ A, const __bf16* __restrict__ Bt,
        float* __restrict__ Out, int N, int K) {
    __shared__ __bf16 lA[2][BM * LDT];   // [m][k]
    __shared__ __bf16 lB[2][BN * LDT];   // [n][k]

    const int t    = threadIdx.x;
    const int lane = t & 31;
    const int wid  = t >> 5;
    const int half = lane >> 4;       // 0: lanes 0-15, 1: lanes 16-31
    const int l16  = lane & 15;
    const int m0   = blockIdx.y * BM;
    const int n0   = blockIdx.x * BN;
    const int wm0  = (wid & 3) * 32;
    const int wn0  = (wid >> 2) * 64;

    // staging: 512 x 16B chunks per tile, 2 per thread (rows r and r+64)
    const int crow = t >> 2, cseg = t & 3;
    const __bf16* gA0 = A  + (size_t)(m0 + crow) * K + cseg * 8;
    const __bf16* gA1 = A  + (size_t)(m0 + crow + 64) * K + cseg * 8;
    const __bf16* gB0 = Bt + (size_t)(n0 + crow) * K + cseg * 8;
    const __bf16* gB1 = Bt + (size_t)(n0 + crow + 64) * K + cseg * 8;
    const int d0 = crow * LDT + cseg * 8;
    const int d1 = (crow + 64) * LDT + cseg * 8;

    v8f acc[2][4];
#pragma unroll
    for (int i = 0; i < 2; ++i)
#pragma unroll
        for (int j = 0; j < 4; ++j) acc[i][j] = v8f{};

    union Frag { v16bf v; uint4 q[2]; };
    auto compute = [&](int b2) {
        // B frags: lane = col n (mod 16); elems j=0..15 <-> k = half*16 + j
        Frag fb[4];
#pragma unroll
        for (int tn = 0; tn < 4; ++tn) {
            int col = wn0 + tn * 16 + l16;
            fb[tn].q[0] = *(const uint4*)&lB[b2][col * LDT + half * 16];
            fb[tn].q[1] = *(const uint4*)&lB[b2][col * LDT + half * 16 + 8];
        }
        // A frags: lane = row m (mod 16); elems j=0..15 <-> k = half*16 + j
#pragma unroll
        for (int tm = 0; tm < 2; ++tm) {
            int row = wm0 + tm * 16 + l16;
            Frag fa;
            fa.q[0] = *(const uint4*)&lA[b2][row * LDT + half * 16];
            fa.q[1] = *(const uint4*)&lA[b2][row * LDT + half * 16 + 8];
#pragma unroll
            for (int tn = 0; tn < 4; ++tn)
                acc[tm][tn] = __builtin_amdgcn_wmma_f32_16x16x32_bf16(
                    false, fa.v, false, fb[tn].v, (short)0, acc[tm][tn], false, false);
        }
    };

    const int ksteps = K / BK;
    int buf = 0;

#ifdef USE_ASYNC_LDS
    auto issue = [&](int b2, int kt) {
        const int ko = kt * BK;
        blru_async_cp16(&lA[b2][d0], gA0 + ko);
        blru_async_cp16(&lA[b2][d1], gA1 + ko);
        blru_async_cp16(&lB[b2][d0], gB0 + ko);
        blru_async_cp16(&lB[b2][d1], gB1 + ko);
    };
    issue(0, 0);
    __builtin_amdgcn_s_wait_asynccnt(0);
    __syncthreads();
    for (int kt = 0; kt < ksteps; ++kt) {
        if (kt + 1 < ksteps) issue(buf ^ 1, kt + 1);   // prefetch next tile
        compute(buf);                                  // overlaps async copies
        __builtin_amdgcn_s_wait_asynccnt(0);
        __syncthreads();
        buf ^= 1;
    }
#else
    // register-staged fallback, same double-buffered schedule
    {
        *(uint4*)&lA[0][d0] = *(const uint4*)gA0;
        *(uint4*)&lA[0][d1] = *(const uint4*)gA1;
        *(uint4*)&lB[0][d0] = *(const uint4*)gB0;
        *(uint4*)&lB[0][d1] = *(const uint4*)gB1;
    }
    __syncthreads();
    for (int kt = 0; kt < ksteps; ++kt) {
        uint4 ra0, ra1, rb0, rb1;
        const bool more = (kt + 1 < ksteps);
        if (more) {
            const int ko = (kt + 1) * BK;
            ra0 = *(const uint4*)(gA0 + ko);
            ra1 = *(const uint4*)(gA1 + ko);
            rb0 = *(const uint4*)(gB0 + ko);
            rb1 = *(const uint4*)(gB1 + ko);
        }
        compute(buf);
        if (more) {
            *(uint4*)&lA[buf ^ 1][d0] = ra0;
            *(uint4*)&lA[buf ^ 1][d1] = ra1;
            *(uint4*)&lB[buf ^ 1][d0] = rb0;
            *(uint4*)&lB[buf ^ 1][d1] = rb1;
        }
        __syncthreads();
        buf ^= 1;
    }
#endif

    // C/D layout: VGPR v -> M = v + 8*half, N = lane&15
#pragma unroll
    for (int tm = 0; tm < 2; ++tm)
#pragma unroll
        for (int tn = 0; tn < 4; ++tn) {
            float* p = Out + (size_t)(m0 + wm0 + tm * 16 + half * 8) * N
                           + (n0 + wn0 + tn * 16 + l16);
#pragma unroll
            for (int v = 0; v < 8; ++v) { *p = acc[tm][tn][v]; p += N; }
        }
}

// ---------------------------------------------------------- blocked scan
// Phase 1: local scan of each 64-step segment (zero init), in place on Bu.
// Bu layout: [b][l][n] re at col n, im at col 512+n. Channels n<256 forward,
// n>=256 backward (time t -> physical l = L-1-t). Input mask (l >= len -> 0).
__global__ __launch_bounds__(256) void blru_scan1_kernel(
        float* __restrict__ Bu, float2* __restrict__ carry,
        const float* __restrict__ lam_re, const float* __restrict__ lam_im,
        const int* __restrict__ lengths) {
    const int n = blockIdx.x * 256 + threadIdx.x;   // 0..511
    const int s = blockIdx.y;                       // 0..63
    const int b = blockIdx.z;                       // 0..7
    const float lr = lam_re[n], li = lam_im[n];
    const int len = lengths[b];
    const bool fwd = (n < (N_DIM / 2));
    const size_t rowbase = (size_t)b * LSEQ;
    float xr = 0.f, xi = 0.f;
    for (int i = 0; i < SEGLEN; ++i) {
        int tt = s * SEGLEN + i;
        int l  = fwd ? tt : (LSEQ - 1 - tt);
        size_t idx = (rowbase + l) * 1024 + n;
        float br = Bu[idx], bi = Bu[idx + 512];
        if (l >= len) { br = 0.f; bi = 0.f; }
        float nr = fmaf(lr, xr, fmaf(-li, xi, br));
        float ni = fmaf(lr, xi, fmaf( li, xr, bi));
        xr = nr; xi = ni;
        Bu[idx] = xr; Bu[idx + 512] = xi;
    }
    carry[((size_t)b * N_DIM + n) * SEGS + s] = make_float2(xr, xi);
}

// Phase 2: per (b,n) scan of segment carries with lam^64 (repeated squaring);
// rewrites carry[s] with the carry-IN for segment s.
__global__ void blru_scan2_kernel(float2* __restrict__ carry,
                                  const float* __restrict__ lam_re,
                                  const float* __restrict__ lam_im) {
    int idx = blockIdx.x * 256 + threadIdx.x;       // < 8*512
    int n = idx & (N_DIM - 1);
    float pr = lam_re[n], pi = lam_im[n];
#pragma unroll
    for (int q = 0; q < 6; ++q) {                   // lam^(2^6) = lam^64
        float nr = pr * pr - pi * pi;
        float ni = 2.f * pr * pi;
        pr = nr; pi = ni;
    }
    float cr = 0.f, ci = 0.f;
    float2* base = &carry[(size_t)idx * SEGS];
    for (int s = 0; s < SEGS; ++s) {
        float2 e = base[s];
        base[s] = make_float2(cr, ci);              // carry-in for segment s
        float nr = fmaf(pr, cr, fmaf(-pi, ci, e.x));
        float ni = fmaf(pr, ci, fmaf( pi, cr, e.y));
        cr = nr; ci = ni;
    }
}

// Phase 3: x_t = x_local_t + lam^(i+1) * carry_in; apply output mask; emit bf16.
__global__ __launch_bounds__(256) void blru_scan3_kernel(
        const float* __restrict__ Bu, __bf16* __restrict__ Xb,
        const float2* __restrict__ carry,
        const float* __restrict__ lam_re, const float* __restrict__ lam_im,
        const int* __restrict__ lengths) {
    const int n = blockIdx.x * 256 + threadIdx.x;
    const int s = blockIdx.y;
    const int b = blockIdx.z;
    const float lr = lam_re[n], li = lam_im[n];
    const int len = lengths[b];
    const bool fwd = (n < (N_DIM / 2));
    const float2 cin = carry[((size_t)b * N_DIM + n) * SEGS + s];
    const size_t rowbase = (size_t)b * LSEQ;
    float pr = lr, pi = li;                         // lam^(i+1)
    for (int i = 0; i < SEGLEN; ++i) {
        int tt = s * SEGLEN + i;
        int l  = fwd ? tt : (LSEQ - 1 - tt);
        size_t idx = (rowbase + l) * 1024 + n;
        float xr = Bu[idx]       + (pr * cin.x - pi * cin.y);
        float xi = Bu[idx + 512] + (pr * cin.y + pi * cin.x);
        if (l >= len) { xr = 0.f; xi = 0.f; }
        Xb[idx]       = f2bf(xr);
        Xb[idx + 512] = f2bf(xi);
        float nr = pr * lr - pi * li;
        float ni = pr * li + pi * lr;
        pr = nr; pi = ni;
    }
}

// ---------------------------------------------------------------------------
extern "C" void kernel_launch(void* const* d_in, const int* in_sizes, int n_in,
                              void* d_out, int out_size, void* d_ws, size_t ws_size,
                              hipStream_t stream) {
    const float* u         = (const float*)d_in[0];   // (8, 4096, 512)
    const int*   lengths   = (const int*)  d_in[1];   // (8,)
    const float* nu_log    = (const float*)d_in[2];   // (512,)
    const float* theta_log = (const float*)d_in[3];   // (512,)
    const float* Bin       = (const float*)d_in[4];   // (512, 512, 2)
    const float* Cin       = (const float*)d_in[5];   // (2, 512, 512)
    float* y = (float*)d_out;                          // (8, 4096, 512)

    char* ws = (char*)d_ws;
    size_t off = 0;
    auto alloc = [&](size_t bytes) -> void* {
        off = (off + 255) & ~(size_t)255;
        void* p = ws + off;
        off += bytes;
        return p;
    };
    float*  lam_re = (float*) alloc((size_t)N_DIM * 4);
    float*  lam_im = (float*) alloc((size_t)N_DIM * 4);
    __bf16* W1t    = (__bf16*)alloc((size_t)1024 * 512 * 2);
    __bf16* W2t    = (__bf16*)alloc((size_t)512 * 1024 * 2);
    float2* carry  = (float2*)alloc((size_t)BSZ * N_DIM * SEGS * 8);
    float*  Bu     = (float*) alloc((size_t)MROWS * 1024 * 4);   // 128 MB
    __bf16* Xb     = (__bf16*)alloc((size_t)MROWS * 1024 * 2);   // 64 MB
    __bf16* uB     = (__bf16*)Xb;   // alias: uB dead before Xb is first written

    blru_lam_kernel<<<2, 256, 0, stream>>>(nu_log, theta_log, lam_re, lam_im);
    blru_w1_kernel<<<2048, 256, 0, stream>>>(Bin, W1t);
    blru_w2_kernel<<<2048, 256, 0, stream>>>(Cin, W2t);
    blru_cvtu_kernel<<<(MROWS * 512 / 4 + 255) / 256, 256, 0, stream>>>(
        u, uB, MROWS * 512 / 4);

    // Bu = uB (32768x512) @ W1 (512x1024)
    blru_gemm_kernel<<<dim3(1024 / BN, MROWS / BM), 256, 0, stream>>>(
        uB, W1t, Bu, 1024, 512);

    blru_scan1_kernel<<<dim3(N_DIM / 256, SEGS, BSZ), 256, 0, stream>>>(
        Bu, carry, lam_re, lam_im, lengths);
    blru_scan2_kernel<<<(BSZ * N_DIM) / 256, 256, 0, stream>>>(carry, lam_re, lam_im);
    blru_scan3_kernel<<<dim3(N_DIM / 256, SEGS, BSZ), 256, 0, stream>>>(
        Bu, Xb, carry, lam_re, lam_im, lengths);

    // y = Xb (32768x1024) @ W2 (1024x512)
    blru_gemm_kernel<<<dim3(512 / BN, MROWS / BM), 256, 0, stream>>>(
        Xb, W2t, y, 512, 1024);
}
